// DDoSDetectionModel_12171937317153
// MI455X (gfx1250) — matrix operand order repaired and checked by reading, not compiled, not run on hardware
//
#include <hip/hip_runtime.h>
#include <hip/hip_bf16.h>

typedef __attribute__((ext_vector_type(16))) __bf16 v16bf;
typedef __attribute__((ext_vector_type(8)))  __bf16 v8bf;
typedef __attribute__((ext_vector_type(8)))  float  v8f;

#define D_MODEL   256
#define D_INNER   1024
#define DT_RANK   16
#define D_STATE   32
#define N_LAYERS  4
#define INPUT_DIM 78

// Packed bf16 weight fragments: fragment (kt,nt) = 32x16 tile in WMMA-B layout.
// Within a fragment: lane l (0..31), j (0..15):
//   elem = W[kt*32 + (j<8 ? (l>>4)*8 + j : 16 + (l>>4)*8 + j-8)][nt*16 + (l&15)]   (zero if k >= K)
#define FRAG 512
#define NT_IN    128
#define KT_IN    8
#define NT_X     5
#define KT_X     32
#define NT_DT    64
#define NT_OUT   16
#define KT_OUT   32
#define NT_PROJ  16
#define KT_PROJ  3
#define HWP_IN   (KT_IN*NT_IN*FRAG)      // 524288 halves / layer
#define HWP_X    (KT_X*NT_X*FRAG)        // 81920
#define HWP_DT   (NT_DT*FRAG)            // 32768 (K padded 16->32)
#define HWP_OUT  (KT_OUT*NT_OUT*FRAG)    // 262144
#define HWP_PROJ (KT_PROJ*NT_PROJ*FRAG)  // 24576 (K padded 78->96)

__global__ __launch_bounds__(256)
void pack_bf16_kernel(const float* __restrict__ src, __bf16* __restrict__ dst,
                      int K, int N, int Kpad) {
    int numNt = N >> 4;
    int total = (Kpad >> 5) * numNt * FRAG;
    for (int idx = blockIdx.x * blockDim.x + threadIdx.x; idx < total;
         idx += gridDim.x * blockDim.x) {
        int frag = idx >> 9;
        int rem  = idx & 511;
        int l    = rem >> 4;
        int j    = rem & 15;
        int kt   = frag / numNt;
        int nt   = frag - kt * numNt;
        int k    = kt * 32 + ((j < 8) ? ((l >> 4) * 8 + j) : (16 + (l >> 4) * 8 + (j - 8)));
        int n    = nt * 16 + (l & 15);
        float v  = (k < K) ? src[k * N + n] : 0.0f;
        dst[idx] = (__bf16)v;
    }
}

// A-fragment (16x32 bf16) gather from row-major bf16 LDS:
// lane l: row m=l&15; halves 0..7 = K[(l>=16?8:0)+j], halves 8..15 = K[16+...]
__device__ __forceinline__ v16bf make_a(const __bf16* rowk, int h8) {
    v8bf lo = *(const v8bf*)(rowk + h8);
    v8bf hi = *(const v8bf*)(rowk + 16 + h8);
    v16bf a;
#pragma unroll
    for (int j = 0; j < 8; j++) { a[j] = lo[j]; a[8 + j] = hi[j]; }
    return a;
}

__device__ __forceinline__ float silu_f(float v) {
    return v / (1.0f + __expf(-v));
}

__global__ __launch_bounds__(256)
void mamba_fused_kernel(const float* __restrict__ x,
                        const float* __restrict__ bproj,
                        const float* __restrict__ normw,
                        const float* __restrict__ convw,
                        const float* __restrict__ convb,
                        const float* __restrict__ bdt,
                        const float* __restrict__ Dw,
                        const float* __restrict__ Wfin,
                        const float* __restrict__ bfin,
                        const __bf16* __restrict__ WpProj,
                        const __bf16* __restrict__ WpIn,
                        const __bf16* __restrict__ WpX,
                        const __bf16* __restrict__ WpDt,
                        const __bf16* __restrict__ WpOut,
                        float* __restrict__ out) {
    __shared__ float  hS[16 * 256];    // residual stream (fp32)
    __shared__ float  dbcS[16 * 80];   // dt/B/C projection output
    __shared__ __bf16 uS[16 * 264];    // x tile (padded), then rmsnorm output
    __shared__ __bf16 xiS[16 * 1032];  // xi, later y (A of GEMM2/4)
    __shared__ float  redS[256];
    __shared__ float  rstdS[16];
    __shared__ float  sS[16];

    const int tid  = threadIdx.x;
    const int lane = tid & 31;
    const int wv   = tid >> 5;         // wave 0..7
    const int m16  = lane & 15;        // A row / B col / D col
    const int h8   = (lane >> 4) * 8;  // A K-offset / D M-offset
    const int tm   = tid >> 4;         // 0..15 (row for reductions)
    const int tj   = tid & 15;
    const int row0 = blockIdx.x * 16;

    // ---- stage x tile as bf16 into uS, zero-padded K: 78 -> 96 ----
    for (int i = tid; i < 16 * 96; i += 256) {
        int m = i / 96, c = i - m * 96;
        uS[m * 264 + c] = (c < INPUT_DIM) ? (__bf16)x[(row0 + m) * INPUT_DIM + c]
                                          : (__bf16)0.0f;
    }
    __syncthreads();

    // ---- GEMM0: h = x @ W_proj_in + b  (16x96pad @ 96x256, WMMA) ----
    {
        v8f accP[2];
#pragma unroll
        for (int t = 0; t < 2; t++)
#pragma unroll
            for (int r = 0; r < 8; r++) accP[t][r] = 0.0f;
#pragma unroll
        for (int kt = 0; kt < KT_PROJ; kt++) {
            v16bf a = make_a(&uS[m16 * 264 + kt * 32], h8);
#pragma unroll
            for (int t = 0; t < 2; t++) {
                v16bf b = *(const v16bf*)(WpProj + (kt * NT_PROJ + wv * 2 + t) * FRAG + lane * 16);
                accP[t] = __builtin_amdgcn_wmma_f32_16x16x32_bf16(
                    false, a, false, b, (short)0, accP[t], false, false);
            }
        }
#pragma unroll
        for (int t = 0; t < 2; t++) {
            int nb = (wv * 2 + t) * 16 + m16;
            float bb = bproj[nb];
#pragma unroll
            for (int r = 0; r < 8; r++)
                hS[(r + h8) * 256 + nb] = accP[t][r] + bb;
        }
    }

    for (int l = 0; l < N_LAYERS; l++) {
        __syncthreads();
        // ---- rmsnorm -> uS (bf16) ----
        float p = 0.0f;
#pragma unroll
        for (int i = 0; i < 16; i++) {
            float v = hS[tm * 256 + tj + 16 * i];
            p += v * v;
        }
        redS[tm * 16 + tj] = p;
        __syncthreads();
        if (tj == 0) {
            float t = 0.0f;
            for (int i = 0; i < 16; i++) t += redS[tm * 16 + i];
            rstdS[tm] = rsqrtf(t * (1.0f / 256.0f) + 1e-5f);
        }
        __syncthreads();
        {
            float rs = rstdS[tm];
            for (int ci = 0; ci < 16; ci++) {
                int c = tj + 16 * ci;
                uS[tm * 264 + c] = (__bf16)(hS[tm * 256 + c] * rs * normw[l * D_MODEL + c]);
            }
        }
        __syncthreads();

        // ---- GEMM1: xz = u @ W_in  (16x256 @ 256x2048)
        // wave wv owns xi cols [wv*128, wv*128+128) and z cols 1024 + same.
        const __bf16* wpin = WpIn + l * HWP_IN;
        v8f accXi[8], accZ[8];
#pragma unroll
        for (int t = 0; t < 8; t++)
#pragma unroll
            for (int r = 0; r < 8; r++) { accXi[t][r] = 0.0f; accZ[t][r] = 0.0f; }
#pragma unroll 2
        for (int kt = 0; kt < 8; kt++) {
            v16bf a = make_a(&uS[m16 * 264 + kt * 32], h8);
#pragma unroll
            for (int t = 0; t < 8; t++) {
                v16bf b1 = *(const v16bf*)(wpin + (kt * NT_IN + (wv * 8 + t)) * FRAG + lane * 16);
                accXi[t] = __builtin_amdgcn_wmma_f32_16x16x32_bf16(
                    false, a, false, b1, (short)0, accXi[t], false, false);
                v16bf b2 = *(const v16bf*)(wpin + (kt * NT_IN + 64 + (wv * 8 + t)) * FRAG + lane * 16);
                accZ[t] = __builtin_amdgcn_wmma_f32_16x16x32_bf16(
                    false, a, false, b2, (short)0, accZ[t], false, false);
            }
        }
        // epilogue: xi = silu(conv-last-tap), z-gate kept in registers
        float zsil[8][8];
#pragma unroll
        for (int t = 0; t < 8; t++) {
            int nb = (wv * 8 + t) * 16 + m16;
            float cw = convw[(l * D_INNER + nb) * 16 + 15];
            float cb = convb[l * D_INNER + nb];
#pragma unroll
            for (int r = 0; r < 8; r++) {
                float v = accXi[t][r] * cw + cb;
                xiS[(r + h8) * 1032 + nb] = (__bf16)silu_f(v);
                zsil[t][r] = silu_f(accZ[t][r]);
            }
        }
        __syncthreads();

        // ---- GEMM2: dbc = xi @ W_x  (16x1024 @ 1024x80), waves 0..4 ----
        if (wv < 5) {
            const __bf16* wpx = WpX + l * HWP_X;
            v8f acc;
#pragma unroll
            for (int r = 0; r < 8; r++) acc[r] = 0.0f;
#pragma unroll 4
            for (int kt = 0; kt < 32; kt++) {
                v16bf a = make_a(&xiS[m16 * 1032 + kt * 32], h8);
                v16bf b = *(const v16bf*)(wpx + (kt * NT_X + wv) * FRAG + lane * 16);
                acc = __builtin_amdgcn_wmma_f32_16x16x32_bf16(
                    false, a, false, b, (short)0, acc, false, false);
            }
#pragma unroll
            for (int r = 0; r < 8; r++)
                dbcS[(r + h8) * 80 + wv * 16 + m16] = acc[r];
        }
        __syncthreads();

        // ---- s[m] = dot(B, C) over D_STATE=32 ----
        {
            float pp = dbcS[tm * 80 + 16 + tj] * dbcS[tm * 80 + 48 + tj]
                     + dbcS[tm * 80 + 32 + tj] * dbcS[tm * 80 + 64 + tj];
            redS[tm * 16 + tj] = pp;
        }
        __syncthreads();
        if (tj == 0) {
            float t = 0.0f;
            for (int i = 0; i < 16; i++) t += redS[tm * 16 + i];
            sS[tm] = t;
        }
        __syncthreads();

        // ---- GEMM3: delta_pre = dbc[:, :16] @ W_dt (K padded to 32) ----
        const __bf16* wpdt = WpDt + l * HWP_DT;
        v16bf adt;
#pragma unroll
        for (int j = 0; j < 8; j++) {
            adt[j]     = (__bf16)dbcS[m16 * 80 + h8 + j];  // K = h8+j < 16
            adt[8 + j] = (__bf16)0.0f;                     // K >= 16: zero pad
        }
        v8f accD[8];
#pragma unroll
        for (int t = 0; t < 8; t++) {
#pragma unroll
            for (int r = 0; r < 8; r++) accD[t][r] = 0.0f;
            v16bf b = *(const v16bf*)(wpdt + (wv * 8 + t) * FRAG + lane * 16);
            accD[t] = __builtin_amdgcn_wmma_f32_16x16x32_bf16(
                false, adt, false, b, (short)0, accD[t], false, false);
        }

        // ---- y = (softplus(delta)*s + D) * xi * silu(z): overwrite xiS ----
#pragma unroll
        for (int t = 0; t < 8; t++) {
            int nb = (wv * 8 + t) * 16 + m16;
            float bd = bdt[l * D_INNER + nb];
            float Dn = Dw[l * D_INNER + nb];
#pragma unroll
            for (int r = 0; r < 8; r++) {
                float dp = accD[t][r] + bd;
                float delta = (dp > 20.0f) ? dp : log1pf(__expf(dp));
                int mr = r + h8;
                float xiv = (float)xiS[mr * 1032 + nb];
                float y = (delta * sS[mr] + Dn) * xiv * zsil[t][r];
                xiS[mr * 1032 + nb] = (__bf16)y;
            }
        }
        __syncthreads();

        // ---- GEMM4: h += y @ W_out  (16x1024 @ 1024x256) ----
        const __bf16* wpo = WpOut + l * HWP_OUT;
        v8f accO[2];
#pragma unroll
        for (int t = 0; t < 2; t++)
#pragma unroll
            for (int r = 0; r < 8; r++) accO[t][r] = 0.0f;
#pragma unroll 4
        for (int kt = 0; kt < 32; kt++) {
            v16bf a = make_a(&xiS[m16 * 1032 + kt * 32], h8);
#pragma unroll
            for (int t = 0; t < 2; t++) {
                v16bf b = *(const v16bf*)(wpo + (kt * NT_OUT + wv * 2 + t) * FRAG + lane * 16);
                accO[t] = __builtin_amdgcn_wmma_f32_16x16x32_bf16(
                    false, a, false, b, (short)0, accO[t], false, false);
            }
        }
#pragma unroll
        for (int t = 0; t < 2; t++) {
            int nb = (wv * 2 + t) * 16 + m16;
#pragma unroll
            for (int r = 0; r < 8; r++)
                hS[(r + h8) * 256 + nb] += accO[t][r];
        }
    }
    __syncthreads();

    // ---- head: sigmoid(h @ W_final + b_final) ----
    {
        float p = 0.0f;
#pragma unroll
        for (int i = 0; i < 16; i++) {
            int c = tj + 16 * i;
            p += hS[tm * 256 + c] * Wfin[c];
        }
        redS[tm * 16 + tj] = p;
        __syncthreads();
        if (tj == 0) {
            float t = bfin[0];
            for (int i = 0; i < 16; i++) t += redS[tm * 16 + i];
            out[row0 + tm] = 1.0f / (1.0f + __expf(-t));
        }
    }
}

extern "C" void kernel_launch(void* const* d_in, const int* in_sizes, int n_in,
                              void* d_out, int out_size, void* d_ws, size_t ws_size,
                              hipStream_t stream) {
    (void)in_sizes; (void)n_in; (void)out_size; (void)ws_size;
    const float* x     = (const float*)d_in[0];
    const float* Wproj = (const float*)d_in[1];
    const float* bproj = (const float*)d_in[2];
    const float* normw = (const float*)d_in[3];
    const float* W_in  = (const float*)d_in[4];
    const float* convw = (const float*)d_in[5];
    const float* convb = (const float*)d_in[6];
    const float* W_x   = (const float*)d_in[7];
    const float* W_dt  = (const float*)d_in[8];
    const float* bdt   = (const float*)d_in[9];
    // d_in[10] = A_log: unused (L==1 => single scan step from h0=0)
    const float* Dw    = (const float*)d_in[11];
    const float* W_out = (const float*)d_in[12];
    const float* Wfin  = (const float*)d_in[13];
    const float* bfin  = (const float*)d_in[14];

    __bf16* WpIn   = (__bf16*)d_ws;
    __bf16* WpX    = WpIn  + 4 * HWP_IN;
    __bf16* WpDt   = WpX   + 4 * HWP_X;
    __bf16* WpOut  = WpDt  + 4 * HWP_DT;
    __bf16* WpProj = WpOut + 4 * HWP_OUT;

    for (int l = 0; l < N_LAYERS; l++) {
        pack_bf16_kernel<<<(HWP_IN + 255) / 256, 256, 0, stream>>>(
            W_in + (size_t)l * 256 * 2048, WpIn + (size_t)l * HWP_IN, 256, 2048, 256);
        pack_bf16_kernel<<<(HWP_X + 255) / 256, 256, 0, stream>>>(
            W_x + (size_t)l * 1024 * 80, WpX + (size_t)l * HWP_X, 1024, 80, 1024);
        pack_bf16_kernel<<<(HWP_DT + 255) / 256, 256, 0, stream>>>(
            W_dt + (size_t)l * 16 * 1024, WpDt + (size_t)l * HWP_DT, 16, 1024, 32);
        pack_bf16_kernel<<<(HWP_OUT + 255) / 256, 256, 0, stream>>>(
            W_out + (size_t)l * 1024 * 256, WpOut + (size_t)l * HWP_OUT, 1024, 256, 1024);
    }
    pack_bf16_kernel<<<(HWP_PROJ + 255) / 256, 256, 0, stream>>>(
        Wproj, WpProj, INPUT_DIM, 256, 96);

    mamba_fused_kernel<<<4096 / 16, 256, 0, stream>>>(
        x, bproj, normw, convw, convb, bdt, Dw, Wfin, bfin,
        WpProj, WpIn, WpX, WpDt, WpOut, (float*)d_out);
}
